// SimpleNet_83064667504978
// MI455X (gfx1250) — compile-verified, matrix operand order checked by reading, not run on hardware
//
#include <hip/hip_runtime.h>
#include <hip/hip_bf16.h>

#define N_NODES 40000
#define N_EDGES 640000
#define N_GRAPHS 64
#define D 128
#define NPANEL (N_NODES / 16)   // 2500 exactly

typedef __attribute__((ext_vector_type(16))) _Float16 v16h;
typedef __attribute__((ext_vector_type(8)))  float    v8f;

// Load one 16x32 f16 A/B fragment element-set for this lane from a row-major
// f32 matrix row. Per ISA layout (16-bit A-matrix 16x32): lane&15 = M (or N for B),
// lane>>4 selects K-octets {hi*8..hi*8+7} and {16+hi*8..16+hi*8+7}.
__device__ __forceinline__ v16h load_frag(const float* __restrict__ row, int kb, int hi, float scale) {
    const float4* p0 = (const float4*)(row + kb + hi * 8);
    const float4* p1 = (const float4*)(row + kb + hi * 8 + 16);
    float4 q0 = p0[0], q1 = p0[1];
    float4 q2 = p1[0], q3 = p1[1];
    v16h f;
    f[0]  = (_Float16)(q0.x * scale); f[1]  = (_Float16)(q0.y * scale);
    f[2]  = (_Float16)(q0.z * scale); f[3]  = (_Float16)(q0.w * scale);
    f[4]  = (_Float16)(q1.x * scale); f[5]  = (_Float16)(q1.y * scale);
    f[6]  = (_Float16)(q1.z * scale); f[7]  = (_Float16)(q1.w * scale);
    f[8]  = (_Float16)(q2.x * scale); f[9]  = (_Float16)(q2.y * scale);
    f[10] = (_Float16)(q2.z * scale); f[11] = (_Float16)(q2.w * scale);
    f[12] = (_Float16)(q3.x * scale); f[13] = (_Float16)(q3.y * scale);
    f[14] = (_Float16)(q3.z * scale); f[15] = (_Float16)(q3.w * scale);
    return f;
}

// out[M,128] = (A1 * s1) @ W1^T + bias (+ A2 @ W2^T) ; optional relu.
// s1 = 1/max(deg[row],1) when deg != nullptr. One wave per 16-row panel.
__global__ void gemm_wmma_kernel(const float* __restrict__ A1, const float* __restrict__ W1,
                                 const float* __restrict__ bias,
                                 const float* __restrict__ A2, const float* __restrict__ W2,
                                 const float* __restrict__ deg,
                                 float* __restrict__ out, int relu) {
    int wave = (blockIdx.x * blockDim.x + threadIdx.x) >> 5;   // wave-uniform
    int lane = threadIdx.x & 31;
    if (wave >= NPANEL) return;                                // uniform branch
    int m0  = wave * 16;
    int mlo = lane & 15;
    int hi  = lane >> 4;
    int mrow = m0 + mlo;

    float s1 = 1.0f;
    if (deg) s1 = 1.0f / fmaxf(deg[mrow], 1.0f);

    // Preload A fragments once per panel (reused across all 8 N-tiles).
    v16h a1[4], a2[4];
#pragma unroll
    for (int kb = 0; kb < 4; ++kb)
        a1[kb] = load_frag(A1 + (size_t)mrow * D, kb * 32, hi, s1);
    if (A2) {
#pragma unroll
        for (int kb = 0; kb < 4; ++kb)
            a2[kb] = load_frag(A2 + (size_t)mrow * D, kb * 32, hi, 1.0f);
    }

    for (int n0 = 0; n0 < 8; ++n0) {
        int ncol = n0 * 16 + mlo;          // lane&15 = N for B/C/D fragments
        float bv = bias[ncol];
        v8f c;
#pragma unroll
        for (int r = 0; r < 8; ++r) c[r] = bv;

#pragma unroll
        for (int kb = 0; kb < 4; ++kb) {
            v16h b1 = load_frag(W1 + (size_t)ncol * D, kb * 32, hi, 1.0f); // B[k][n] = W[n][k]
            c = __builtin_amdgcn_wmma_f32_16x16x32_f16(false, a1[kb], false, b1,
                                                       (short)0, c, false, false);
        }
        if (A2) {
#pragma unroll
            for (int kb = 0; kb < 4; ++kb) {
                v16h b2 = load_frag(W2 + (size_t)ncol * D, kb * 32, hi, 1.0f);
                c = __builtin_amdgcn_wmma_f32_16x16x32_f16(false, a2[kb], false, b2,
                                                           (short)0, c, false, false);
            }
        }
        // D layout: VGPR r -> M = m0 + r + hi*8, N = ncol
#pragma unroll
        for (int r = 0; r < 8; ++r) {
            float v = c[r];
            if (relu) v = fmaxf(v, 0.0f);
            out[(size_t)(m0 + hi * 8 + r) * D + ncol] = v;
        }
    }
}

// One wave per edge: gather h[src] (float4/lane), atomic scatter-add into agg[dst].
__global__ void edge_scatter_kernel(const float* __restrict__ h, const int* __restrict__ ei,
                                    float* __restrict__ agg, float* __restrict__ deg) {
    long long t = (long long)blockIdx.x * blockDim.x + threadIdx.x;
    int e    = (int)(t >> 5);
    int lane = (int)(t & 31);
    if (e >= N_EDGES) return;
    int src = ei[e];
    int dst = ei[N_EDGES + e];
    float4 v = ((const float4*)(h + (size_t)src * D))[lane];
    float* ap = agg + (size_t)dst * D + lane * 4;
    atomicAdd(ap + 0, v.x);
    atomicAdd(ap + 1, v.y);
    atomicAdd(ap + 2, v.z);
    atomicAdd(ap + 3, v.w);
    if (lane == 0) atomicAdd(deg + dst, 1.0f);
}

// In-place LayerNorm (population variance) + ReLU. One wave per row.
__global__ void ln_relu_kernel(float* __restrict__ h, const float* __restrict__ g,
                               const float* __restrict__ b) {
    int wave = (blockIdx.x * blockDim.x + threadIdx.x) >> 5;
    int lane = threadIdx.x & 31;
    if (wave >= N_NODES) return;
    float* row = h + (size_t)wave * D;
    float4 v = ((const float4*)row)[lane];
    float s  = v.x + v.y + v.z + v.w;
    float s2 = v.x * v.x + v.y * v.y + v.z * v.z + v.w * v.w;
#pragma unroll
    for (int off = 16; off >= 1; off >>= 1) {
        s  += __shfl_xor(s,  off, 32);
        s2 += __shfl_xor(s2, off, 32);
    }
    float mean = s * (1.0f / 128.0f);
    float var  = s2 * (1.0f / 128.0f) - mean * mean;
    float rstd = rsqrtf(var + 1e-5f);
    float4 gg = ((const float4*)g)[lane];
    float4 bb = ((const float4*)b)[lane];
    float4 o;
    o.x = fmaxf((v.x - mean) * rstd * gg.x + bb.x, 0.0f);
    o.y = fmaxf((v.y - mean) * rstd * gg.y + bb.y, 0.0f);
    o.z = fmaxf((v.z - mean) * rstd * gg.z + bb.z, 0.0f);
    o.w = fmaxf((v.w - mean) * rstd * gg.w + bb.w, 0.0f);
    ((float4*)row)[lane] = o;
}

// One wave per node: atomic segment-sum of node embeddings into graph rows.
__global__ void graph_pool_kernel(const float* __restrict__ node, const int* __restrict__ batch,
                                  float* __restrict__ gout) {
    int wave = (blockIdx.x * blockDim.x + threadIdx.x) >> 5;
    int lane = threadIdx.x & 31;
    if (wave >= N_NODES) return;
    int gi = batch[wave];
    float4 v = ((const float4*)(node + (size_t)wave * D))[lane];
    float* gp = gout + (size_t)gi * D + lane * 4;
    atomicAdd(gp + 0, v.x);
    atomicAdd(gp + 1, v.y);
    atomicAdd(gp + 2, v.z);
    atomicAdd(gp + 3, v.w);
}

__global__ void fill_zero_kernel(float* __restrict__ p, long long n) {
    long long i = (long long)blockIdx.x * blockDim.x + threadIdx.x;
    if (i < n) p[i] = 0.0f;
}

extern "C" void kernel_launch(void* const* d_in, const int* in_sizes, int n_in,
                              void* d_out, int out_size, void* d_ws, size_t ws_size,
                              hipStream_t stream) {
    (void)in_sizes; (void)n_in; (void)out_size; (void)ws_size;
    const float* x     = (const float*)d_in[0];
    const int*   ei    = (const int*)d_in[1];
    const int*   batch = (const int*)d_in[2];
    const float* fc_w  = (const float*)d_in[3];
    const float* fc_b  = (const float*)d_in[4];
    const float* Wl    = (const float*)d_in[5];
    const float* bl    = (const float*)d_in[6];
    const float* Wr    = (const float*)d_in[7];
    const float* ln_g  = (const float*)d_in[8];
    const float* ln_b  = (const float*)d_in[9];

    float* node = (float*)d_out;                       // [40000,128]
    float* gout = node + (size_t)N_NODES * D;          // [64,128]

    const size_t NF = (size_t)N_NODES * D;
    float* hA  = (float*)d_ws;
    float* hB  = hA + NF;
    float* agg = hB + NF;
    float* deg = agg + NF;                             // contiguous after agg

    const int BLK = 256;
    const int gemm_grid  = (NPANEL * 32 + BLK - 1) / BLK;       // 313
    const int wavegrid_N = (N_NODES * 32 + BLK - 1) / BLK;      // 5000
    const int edge_grid  = (int)(((long long)N_EDGES * 32) / BLK); // 80000

    // h0 = relu(x @ fc_w^T + fc_b)
    gemm_wmma_kernel<<<gemm_grid, BLK, 0, stream>>>(x, fc_w, fc_b,
                                                    nullptr, nullptr, nullptr, hA, 1);

    const float* hin = hA;
    for (int i = 0; i < 3; ++i) {
        long long fillN = (long long)NF + N_NODES;      // agg + deg together
        fill_zero_kernel<<<(int)((fillN + BLK - 1) / BLK), BLK, 0, stream>>>(agg, fillN);
        edge_scatter_kernel<<<edge_grid, BLK, 0, stream>>>(hin, ei, agg, deg);
        float* out = (i == 2) ? node : ((i == 0) ? hB : hA);
        gemm_wmma_kernel<<<gemm_grid, BLK, 0, stream>>>(agg, Wl + (size_t)i * D * D, bl + i * D,
                                                        hin, Wr + (size_t)i * D * D, deg, out, 0);
        if (i < 2)
            ln_relu_kernel<<<wavegrid_N, BLK, 0, stream>>>(out, ln_g + i * D, ln_b + i * D);
        hin = out;
    }

    fill_zero_kernel<<<(N_GRAPHS * D + BLK - 1) / BLK, BLK, 0, stream>>>(gout, N_GRAPHS * D);
    graph_pool_kernel<<<wavegrid_N, BLK, 0, stream>>>(node, batch, gout);
}
